// Hgru2_30331059045035
// MI455X (gfx1250) — matrix operand order repaired and checked by reading, not compiled, not run on hardware
//
#include <hip/hip_runtime.h>

typedef __attribute__((ext_vector_type(16))) __bf16 v16bf;
typedef __attribute__((ext_vector_type(8)))  float  v8f;
typedef __attribute__((ext_vector_type(4)))  unsigned int v4u;
typedef __attribute__((ext_vector_type(8)))  int  v8i;
typedef __attribute__((ext_vector_type(4)))  int  v4i;

#define TM 128
#define TN 64
#define TK 32
#define ASTRIDE 34   // 32 bf16 + 1 dword pad (matches TDM pad_interval=16dw, pad_amount=1dw)
#define BSTRIDE 34

#define EPI_NONE 0
#define EPI_SILU 1
#define EPI_GATE 2

#if defined(__has_builtin)
# if __has_builtin(__builtin_amdgcn_tensor_load_to_lds) && __has_builtin(__builtin_amdgcn_s_wait_tensorcnt)
#  define USE_TDM 1
# endif
#endif
#ifndef USE_TDM
# define USE_TDM 0
#endif

__device__ __forceinline__ unsigned short f2bf(float f) {
    unsigned int u = __float_as_uint(f);
    u += 0x7fffu + ((u >> 16) & 1u);      // round-to-nearest-even
    return (unsigned short)(u >> 16);
}

__device__ __forceinline__ float sigmoidf_fast(float z) {
#if defined(__gfx1250__)
    return __builtin_amdgcn_rcpf(1.0f + __expf(-z));   // v_exp_f32 + v_rcp_f32
#else
    return 1.0f / (1.0f + __expf(-z));
#endif
}

#if USE_TDM
// Issue a 2D TDM tile load: 32 x tile_d1 elements (2B each) from a row-major
// tensor with row length 1024 elements, into LDS with 1-dword pad per 16 dwords.
__device__ __forceinline__ void tdm_load_tile(unsigned lds_off, const void* gtile,
                                              unsigned tensor_d1, unsigned tile_d1)
{
    unsigned long long ga = (unsigned long long)(size_t)gtile;
    v4u g0;
    g0.x = 1u;                                         // count=1, user descriptor
    g0.y = lds_off;                                    // lds_addr (bytes)
    g0.z = (unsigned)ga;                               // global_addr[31:0]
    g0.w = ((unsigned)(ga >> 32) & 0x01FFFFFFu) | (2u << 30);  // addr[56:32] | type=2
    v8i g1;
    // data_size=1(2B)<<16 | pad_enable<<20 | pad_interval=3(16dw)<<22 | pad_amount=0(1dw)
    g1[0] = (int)0x00D10000u;
    g1[1] = (int)(1024u << 16);                        // tensor_dim0[15:0]=1024 -> bits63:48
    g1[2] = (int)(tensor_d1 << 16);                    // dim0 hi=0 | tensor_dim1[15:0]
    g1[3] = (int)(32u << 16);                          // dim1 hi=0 | tile_dim0=32
    g1[4] = (int)tile_d1;                              // tile_dim1 | tile_dim2=0
    g1[5] = (int)1024u;                                // tensor_dim0_stride[31:0]
    g1[6] = 0;                                         // stride hi / dim1_stride lo
    g1[7] = 0;
    v4i z4 = {0, 0, 0, 0};
    v8i z8 = {0, 0, 0, 0, 0, 0, 0, 0};
    __builtin_amdgcn_tensor_load_to_lds(g0, g1, z4, z4, z8, 0);
}
#endif

// C = act(X @ W^T). X:[M,K] bf16 row-major, W:[Nout,K] bf16 row-major.
// bf16 WMMA, fp32 accumulate. Block tile 128x64, 8 waves, wave tile 32x32 (2x2 WMMA).
// Tiles staged to LDS by the Tensor Data Mover, double buffered.
template <int EPI>
__global__ __launch_bounds__(256)
void gemm_xwt(const unsigned short* __restrict__ X, const unsigned short* __restrict__ W,
              float* __restrict__ out0, float* __restrict__ out1,
              int M, int K, int Nout)
{
    __shared__ unsigned short Asu[2][TM * ASTRIDE];
    __shared__ unsigned short Bsu[2][TN * BSTRIDE];

    const int tid   = threadIdx.x;
    const int m0    = blockIdx.x * TM;
    const int n0    = blockIdx.y * TN;
    const int lane  = tid & 31;
    const int wave  = tid >> 5;
    const int waveM = wave & 3;      // 0..3 -> 32 rows each
    const int waveN = wave >> 2;     // 0..1 -> 32 cols each
    const int half  = lane >> 4;     // 0/1
    const int l16   = lane & 15;

    v8f acc[2][2] = {};

    const int nk = K / TK;

#if USE_TDM
    if (wave == 0) {
        tdm_load_tile((unsigned)(size_t)&Asu[0][0], X + (size_t)m0 * K, (unsigned)M, TM);
        tdm_load_tile((unsigned)(size_t)&Bsu[0][0], W + (size_t)n0 * K, (unsigned)Nout, TN);
    }
#endif

    for (int kt = 0; kt < nk; ++kt) {
        const int cur = kt & 1;
        const int k0  = kt * TK;

#if USE_TDM
        if (wave == 0) __builtin_amdgcn_s_wait_tensorcnt(0);
        __syncthreads();                         // tile `cur` visible to all waves
        if (wave == 0 && kt + 1 < nk) {          // prefetch next tile into other buffer
            int k1 = k0 + TK;
            tdm_load_tile((unsigned)(size_t)&Asu[cur ^ 1][0], X + (size_t)m0 * K + k1, (unsigned)M, TM);
            tdm_load_tile((unsigned)(size_t)&Bsu[cur ^ 1][0], W + (size_t)n0 * K + k1, (unsigned)Nout, TN);
        }
#else
        {   // cooperative bf16 staging fallback (uint = 2 bf16)
            const unsigned int* Xg = (const unsigned int*)X;
            const unsigned int* Wg = (const unsigned int*)W;
            unsigned int* Al = (unsigned int*)&Asu[cur][0];
            unsigned int* Bl = (unsigned int*)&Bsu[cur][0];
            for (int idx = tid; idx < TM * TK / 2; idx += 256) {
                int row = idx >> 4, cu = idx & 15;
                Al[row * (ASTRIDE / 2) + cu] = Xg[((size_t)(m0 + row) * K + k0) / 2 + cu];
            }
            for (int idx = tid; idx < TN * TK / 2; idx += 256) {
                int row = idx >> 4, cu = idx & 15;
                Bl[row * (BSTRIDE / 2) + cu] = Wg[((size_t)(n0 + row) * K + k0) / 2 + cu];
            }
            __syncthreads();
        }
#endif

        // ---- assemble fragments per ISA 16-bit A(16x32)/B(32x16) layouts ----
        union Frag { v16bf v; unsigned int u[8]; };
        Frag afr[2], bfr[2];
        const unsigned int* Abase = (const unsigned int*)&Asu[cur][0];
        const unsigned int* Bbase = (const unsigned int*)&Bsu[cur][0];
#pragma unroll
        for (int mt = 0; mt < 2; ++mt) {
            int mrow = waveM * 32 + mt * 16 + l16;
#pragma unroll
            for (int j = 0; j < 8; ++j) {
                int kk = (j < 4) ? (8 * half + 2 * j)
                                 : (16 + 8 * half + 2 * (j - 4));
                afr[mt].u[j] = Abase[(mrow * ASTRIDE + kk) >> 1];
            }
        }
#pragma unroll
        for (int nt = 0; nt < 2; ++nt) {
            int ncol = waveN * 32 + nt * 16 + l16;
#pragma unroll
            for (int j = 0; j < 8; ++j) {
                int kk = (j < 4) ? (8 * half + 2 * j)
                                 : (16 + 8 * half + 2 * (j - 4));
                bfr[nt].u[j] = Bbase[(ncol * BSTRIDE + kk) >> 1];
            }
        }

#pragma unroll
        for (int mt = 0; mt < 2; ++mt)
#pragma unroll
            for (int nt = 0; nt < 2; ++nt)
                acc[mt][nt] = __builtin_amdgcn_wmma_f32_16x16x32_bf16(
                    false, afr[mt].v, false, bfr[nt].v,
                    (short)0, acc[mt][nt], false, false);

        __syncthreads();                         // all reads of `cur` done before overwrite
    }

    // ---- epilogue: C/D layout: VGPR r -> M = r + 8*half; lane%16 -> N ----
#pragma unroll
    for (int mt = 0; mt < 2; ++mt) {
#pragma unroll
        for (int nt = 0; nt < 2; ++nt) {
            int col = n0 + waveN * 32 + nt * 16 + l16;
#pragma unroll
            for (int r = 0; r < 8; ++r) {
                int row = m0 + waveM * 32 + mt * 16 + r + 8 * half;
                float z = acc[mt][nt][r];
                size_t idx = (size_t)row * Nout + col;
                if (EPI == EPI_SILU) {
                    out0[idx] = z * sigmoidf_fast(z);
                } else if (EPI == EPI_GATE) {
                    float f = sigmoidf_fast(z);        // f = sigmoid(k_proj)
                    out0[idx] = 1.0f - f;              // k = 1 - f
                    out1[idx] = sigmoidf_fast(f);      // a = exp(logsigmoid(f)) = sigmoid(f)
                } else {
                    out0[idx] = z;
                }
            }
        }
    }
}

// fp32 -> bf16 bulk convert (4 elems/thread)
__global__ __launch_bounds__(256)
void to_bf16(const float* __restrict__ x, unsigned short* __restrict__ y, int n)
{
    int i4 = blockIdx.x * 256 + threadIdx.x;
    if (i4 * 4 < n) {
        float4 f = ((const float4*)x)[i4];
        uint2 p;
        p.x = (unsigned int)f2bf(f.x) | ((unsigned int)f2bf(f.y) << 16);
        p.y = (unsigned int)f2bf(f.z) | ((unsigned int)f2bf(f.w) << 16);
        ((uint2*)y)[i4] = p;
    }
}

// GLA scan: S = diag(a) S + k v^T ; o = q^T S. One block per (b, h, 32-col chunk).
#define HD 128
#define VC 32
__global__ __launch_bounds__(256)
void gla_scan(const float* __restrict__ q, const float* __restrict__ k,
              const float* __restrict__ a, const float* __restrict__ v,
              float* __restrict__ o, int Bn, int Nn, int H)
{
    __shared__ float sq[HD], sk[HD], sa[HD], sv[VC], sp[8 * VC];

    const int chunks = HD / VC;                 // 4
    const int c = blockIdx.x % chunks;
    const int h = (blockIdx.x / chunks) % H;
    const int b = blockIdx.x / (chunks * H);
    const int tid = threadIdx.x;
    const int col = tid & (VC - 1);
    const int grp = tid >> 5;                   // 0..7
    const int rbase = grp * 16;
    const int D = H * HD;

    float S[16];
#pragma unroll
    for (int i = 0; i < 16; ++i) S[i] = 0.0f;

    const size_t baseRow = (size_t)b * Nn * D + (size_t)h * HD;

    for (int t = 0; t < Nn; ++t) {
        size_t ro = baseRow + (size_t)t * D;
        if (tid < HD) {
            sq[tid] = q[ro + tid];
            sk[tid] = k[ro + tid];
            sa[tid] = a[ro + tid];
        } else if (tid < HD + VC) {
            sv[tid - HD] = v[ro + c * VC + (tid - HD)];
        }
        __syncthreads();

        float vc = sv[col];
        float sum = 0.0f;
#pragma unroll
        for (int i = 0; i < 16; ++i) {
            int r = rbase + i;
            float s = sa[r] * S[i] + sk[r] * vc;
            S[i] = s;
            sum += sq[r] * s;
        }
        sp[grp * VC + col] = sum;
        __syncthreads();

        if (tid < VC) {
            float oo = 0.0f;
#pragma unroll
            for (int g2 = 0; g2 < 8; ++g2) oo += sp[g2 * VC + tid];
            o[ro + c * VC + tid] = oo;
        }
    }
}

// RMSNorm rows of length D=1024 (fp32 in) -> bf16 out. One block per row.
__global__ __launch_bounds__(256)
void rmsnorm_bf16(const float* __restrict__ x, const float* __restrict__ w,
                  unsigned short* __restrict__ y, int D)
{
    __shared__ float red[8];
    __shared__ float rs;
    const int tid = threadIdx.x;
    const float* p = x + (size_t)blockIdx.x * D;

    float4 f = ((const float4*)p)[tid];
    float ss = f.x * f.x + f.y * f.y + f.z * f.z + f.w * f.w;
#pragma unroll
    for (int off = 16; off > 0; off >>= 1) ss += __shfl_down(ss, off, 32);
    if ((tid & 31) == 0) red[tid >> 5] = ss;
    __syncthreads();
    if (tid == 0) {
        float tot = 0.0f;
#pragma unroll
        for (int i = 0; i < 8; ++i) tot += red[i];
        rs = rsqrtf(tot / (float)D + 1e-6f);
    }
    __syncthreads();
    float r = rs;
    float4 wv = ((const float4*)w)[tid];
    uint2 pk;
    pk.x = (unsigned int)f2bf(f.x * r * wv.x) | ((unsigned int)f2bf(f.y * r * wv.y) << 16);
    pk.y = (unsigned int)f2bf(f.z * r * wv.z) | ((unsigned int)f2bf(f.w * r * wv.w) << 16);
    ((uint2*)(y + (size_t)blockIdx.x * D))[tid] = pk;
}

extern "C" void kernel_launch(void* const* d_in, const int* in_sizes, int n_in,
                              void* d_out, int out_size, void* d_ws, size_t ws_size,
                              hipStream_t stream)
{
    const float* x     = (const float*)d_in[0];
    const float* Wq    = (const float*)d_in[1];
    const float* Wk    = (const float*)d_in[2];
    const float* Wv    = (const float*)d_in[3];
    const float* Wo    = (const float*)d_in[4];
    const float* normw = (const float*)d_in[5];

    const int Bn = 4, Nn = 2048, D = 1024, H = 8;
    const int M = Bn * Nn;                        // 8192
    const size_t per = (size_t)M * D;             // 8.4M elems
    const size_t wsz = (size_t)D * D;             // 1M elems

    char* base = (char*)d_ws;
    float* qb = (float*)base;
    float* kb = qb + per;
    float* ab = kb + per;
    float* vb = ab + per;
    float* ob = vb + per;
    unsigned short* xb  = (unsigned short*)(ob + per);
    unsigned short* wqb = xb + per;
    unsigned short* wkb = wqb + wsz;
    unsigned short* wvb = wkb + wsz;
    unsigned short* wob = wvb + wsz;
    unsigned short* nb  = wob + wsz;

    dim3 bb(256);
    // one-time bf16 downconverts (x reused by 3 GEMMs)
    to_bf16<<<dim3((int)(per / 1024)), bb, 0, stream>>>(x, xb, (int)per);
    to_bf16<<<dim3((int)(wsz / 1024)), bb, 0, stream>>>(Wq, wqb, (int)wsz);
    to_bf16<<<dim3((int)(wsz / 1024)), bb, 0, stream>>>(Wk, wkb, (int)wsz);
    to_bf16<<<dim3((int)(wsz / 1024)), bb, 0, stream>>>(Wv, wvb, (int)wsz);
    to_bf16<<<dim3((int)(wsz / 1024)), bb, 0, stream>>>(Wo, wob, (int)wsz);

    dim3 gg(M / TM, D / TN);
    gemm_xwt<EPI_SILU><<<gg, bb, 0, stream>>>(xb, wqb, qb, nullptr, M, D, D);
    gemm_xwt<EPI_GATE><<<gg, bb, 0, stream>>>(xb, wkb, kb, ab, M, D, D);
    gemm_xwt<EPI_NONE><<<gg, bb, 0, stream>>>(xb, wvb, vb, nullptr, M, D, D);

    gla_scan<<<dim3(Bn * H * (HD / VC)), bb, 0, stream>>>(qb, kb, ab, vb, ob, Bn, Nn, H);

    rmsnorm_bf16<<<dim3(M), bb, 0, stream>>>(ob, normw, nb, D);

    gemm_xwt<EPI_NONE><<<gg, bb, 0, stream>>>(nb, wob, (float*)d_out, nullptr, M, D, D);
}
